// Head_70978629533926
// MI455X (gfx1250) — matrix operand order repaired and checked by reading, not compiled, not run on hardware
//
#include <hip/hip_runtime.h>

#define B_  4
#define T_  4096
#define C_  1024
#define H_  64

typedef __attribute__((ext_vector_type(16))) __bf16 v16bf;
typedef __attribute__((ext_vector_type(8)))  __bf16 v8bf;
typedef __attribute__((ext_vector_type(8)))  float  v8f;

__device__ __forceinline__ v16bf cat8(v8bf lo, v8bf hi) {
  v16bf a;
#pragma unroll
  for (int i = 0; i < 8; ++i) { a[i] = lo[i]; a[8 + i] = hi[i]; }
  return a;
}

// ---------------------------------------------------------------------------
// Kernel 1: W (C_ x H_) fp32 row-major  ->  Wt (H_ x C_) bf16 row-major
// ---------------------------------------------------------------------------
__global__ void __launch_bounds__(256) transpose_w_kernel(const float* __restrict__ W,
                                                          __bf16* __restrict__ Wt) {
  int idx = blockIdx.x * 256 + threadIdx.x;   // 0 .. C_*H_-1
  int k = idx >> 6;                           // / H_
  int n = idx & 63;                           // % H_
  Wt[n * C_ + k] = (__bf16)W[idx];
}

// ---------------------------------------------------------------------------
// Helpers for the software-pipelined QKV GEMM.
// ---------------------------------------------------------------------------
__device__ __forceinline__ v16bf loadA_x(const float* __restrict__ xr) {
  // Per-lane A chunk (contiguous order): [16h..16h+15].
  float buf[16];
  *(float4*)&buf[0]  = ((const float4*)xr)[0];
  *(float4*)&buf[4]  = ((const float4*)xr)[1];
  *(float4*)&buf[8]  = ((const float4*)xr)[2];
  *(float4*)&buf[12] = ((const float4*)xr)[3];
  v16bf a;
#pragma unroll
  for (int i = 0; i < 16; ++i) a[i] = (__bf16)buf[i];
  return a;
}

__device__ __forceinline__ void loadB12(v16bf* __restrict__ Bf,
                                        const __bf16* __restrict__ Wtq,
                                        const __bf16* __restrict__ Wtk,
                                        const __bf16* __restrict__ Wtv,
                                        int nl, int half, int kc) {
#pragma unroll
  for (int t = 0; t < 4; ++t) {
    const int boff = (16 * t + nl) * C_ + kc + 16 * half;
    Bf[t]     = *(const v16bf*)(Wtq + boff);
    Bf[4 + t] = *(const v16bf*)(Wtk + boff);
    Bf[8 + t] = *(const v16bf*)(Wtv + boff);
  }
}

__device__ __forceinline__ void mmas12(v8f* accQ, v8f* accK, v8f* accV,
                                       v16bf a, const v16bf* __restrict__ Bf) {
#pragma unroll
  for (int t = 0; t < 4; ++t) {
    accQ[t] = __builtin_amdgcn_wmma_f32_16x16x32_bf16(false, a, false, Bf[t],     (short)0, accQ[t], false, false);
    accK[t] = __builtin_amdgcn_wmma_f32_16x16x32_bf16(false, a, false, Bf[4 + t], (short)0, accK[t], false, false);
    accV[t] = __builtin_amdgcn_wmma_f32_16x16x32_bf16(false, a, false, Bf[8 + t], (short)0, accV[t], false, false);
  }
}

// ---------------------------------------------------------------------------
// Kernel 2: fused QKV projection (bf16 WMMA, fp32 accum) + RoPE.
// Each wave computes 16 rows x 64 cols of Q, K, V. Q is pre-scaled by 1/8.
// The k-loop is software-pipelined: next A+B fragments are loaded into
// registers while the current 12 WMMAs execute, so WMMAs do not stall on L2.
// ---------------------------------------------------------------------------
__global__ void __launch_bounds__(256) qkv_rope_kernel(
    const float* __restrict__ x,      // (B_*T_, C_)
    const float* __restrict__ cosb,   // (T_, H_)
    const float* __restrict__ sinb,   // (T_, H_)
    const __bf16* __restrict__ Wtq,   // (H_, C_)
    const __bf16* __restrict__ Wtk,
    const __bf16* __restrict__ Wtv,
    __bf16* __restrict__ Qb,          // (B_*T_, H_)
    __bf16* __restrict__ Kb,
    __bf16* __restrict__ Vb) {
  const int lane = threadIdx.x & 31;
  const int wave = threadIdx.x >> 5;
  const int half = lane >> 4;
  const int nl   = lane & 15;
  const int rowBase = blockIdx.x * 128 + wave * 16;   // global row in [0, B_*T_)

  v8f accQ[4] = {v8f{}, v8f{}, v8f{}, v8f{}};
  v8f accK[4] = {v8f{}, v8f{}, v8f{}, v8f{}};
  v8f accV[4] = {v8f{}, v8f{}, v8f{}, v8f{}};

  const float* xrow = x + (rowBase + nl) * C_ + 16 * half;

  // Prologue: fragments for kc = 0.
  v16bf a = loadA_x(xrow);
  v16bf Bf[12];
  loadB12(Bf, Wtq, Wtk, Wtv, nl, half, 0);

  for (int kc = 0; kc < C_ - 32; kc += 32) {
    // Issue next iteration's loads first, then compute with current regs.
    v16bf an = loadA_x(xrow + kc + 32);
    v16bf Bn[12];
    loadB12(Bn, Wtq, Wtk, Wtv, nl, half, kc + 32);

    mmas12(accQ, accK, accV, a, Bf);

    a = an;
#pragma unroll
    for (int i = 0; i < 12; ++i) Bf[i] = Bn[i];
  }
  // Epilogue: last k-step.
  mmas12(accQ, accK, accV, a, Bf);

  // RoPE + store. C layout: VGPR r -> row M = r + 8*half, lane -> col 16t+nl.
  // Pair partner (d ^ 1) lives in lane ^ 1 (same half, same row set).
#pragma unroll
  for (int r = 0; r < 8; ++r) {
    const int row  = rowBase + r + 8 * half;
    const int tpos = row & (T_ - 1);
#pragma unroll
    for (int t = 0; t < 4; ++t) {
      const int d = 16 * t + nl;
      const float c = cosb[tpos * H_ + d];
      const float s = sinb[tpos * H_ + d];
      const float q = accQ[t][r];
      const float k = accK[t][r];
      const float qp = __shfl_xor(q, 1, 32);
      const float kp = __shfl_xor(k, 1, 32);
      const float sgn = (d & 1) ? 1.0f : -1.0f;
      const float qr = q * c + sgn * qp * s;
      const float kr = k * c + sgn * kp * s;
      Qb[row * H_ + d] = (__bf16)(qr * 0.125f);   // fold HEAD^-0.5
      Kb[row * H_ + d] = (__bf16)kr;
      Vb[row * H_ + d] = (__bf16)accV[t][r];
    }
  }
}

// ---------------------------------------------------------------------------
// Kernel 3: causal flash attention. 4 waves/block, 16 query rows per wave,
// 32-key tiles staged in LDS (K row-major, V transposed d-major). Next tile
// is prefetched (global_prefetch_b8) while the current one is computed.
// ---------------------------------------------------------------------------
__global__ void __launch_bounds__(128) attn_kernel(
    const __bf16* __restrict__ Qb,
    const __bf16* __restrict__ Kb,
    const __bf16* __restrict__ Vb,
    float* __restrict__ out) {
  __shared__ __align__(32) __bf16 Kt[32 * 64];      // [key][d]
  __shared__ __align__(32) __bf16 Vt[64 * 32];      // [d][key]
  __shared__ __align__(32) __bf16 Pbuf[4][16 * 32]; // per-wave P, [row][key]

  const int lane = threadIdx.x & 31;
  const int wave = threadIdx.x >> 5;
  const int half = lane >> 4;
  const int nl   = lane & 15;
  const int b      = blockIdx.y;
  const int qblock = blockIdx.x;                 // 64 query rows per block
  const int base   = b * T_;
  const int q0     = qblock * 64 + wave * 16;    // wave's first query (in-batch)

  // Q A-fragments for the two 32-wide K-chunks of HEAD=64.
  v16bf qa[2];
  {
    const __bf16* qr = Qb + (base + q0 + nl) * H_;
#pragma unroll
    for (int kc = 0; kc < 2; ++kc) {
      const __bf16* p = qr + kc * 32 + 16 * half;
      qa[kc] = cat8(*(const v8bf*)p, *(const v8bf*)(p + 8));
    }
  }

  float m_run[8], l_run[8];
#pragma unroll
  for (int r = 0; r < 8; ++r) { m_run[r] = -1.0e30f; l_run[r] = 0.0f; }
  v8f O[4] = {v8f{}, v8f{}, v8f{}, v8f{}};

  const int n_ktiles = 2 * qblock + 2;           // covers keys 0 .. qblock*64+63

  for (int kt = 0; kt < n_ktiles; ++kt) {
    const int k0 = kt * 32;
    __syncthreads();
    {
      // Cooperative tile load: 2048 bf16 per matrix, 16 per thread.
      const int e   = threadIdx.x * 16;
      const int key = e >> 6;
      const int d   = e & 63;
      const __bf16* ks = Kb + (base + k0 + key) * H_ + d;
      *(v8bf*)(&Kt[e])     = *(const v8bf*)ks;
      *(v8bf*)(&Kt[e + 8]) = *(const v8bf*)(ks + 8);
      const __bf16* vs = Vb + (base + k0 + key) * H_ + d;
      v8bf v0 = *(const v8bf*)vs;
      v8bf v1 = *(const v8bf*)(vs + 8);
#pragma unroll
      for (int i = 0; i < 8; ++i) Vt[(d + i) * 32 + key]     = v0[i];
#pragma unroll
      for (int i = 0; i < 8; ++i) Vt[(d + 8 + i) * 32 + key] = v1[i];

      // Prefetch next tile's K/V lines into cache while we compute this one.
      if (kt + 1 < n_ktiles) {
        __builtin_prefetch(ks + 32 * H_, 0, 3);
        __builtin_prefetch(vs + 32 * H_, 0, 3);
      }
    }
    __syncthreads();

    if (k0 <= q0 + 15) {   // wave-uniform: wave has unmasked work in this tile
      // S = Q K^T : two 16x16 tiles (keys k0..+15, k0+16..+31)
      v8f S[2];
#pragma unroll
      for (int sub = 0; sub < 2; ++sub) {
        v8f acc = v8f{};
#pragma unroll
        for (int kc = 0; kc < 2; ++kc) {
          const __bf16* p = &Kt[(sub * 16 + nl) * 64 + kc * 32 + 16 * half];
          v16bf bf = *(const v16bf*)p;
          acc = __builtin_amdgcn_wmma_f32_16x16x32_bf16(false, qa[kc], false, bf, (short)0, acc, false, false);
        }
        S[sub] = acc;
      }

      // Causal mask + online softmax (row = r + 8*half, key col = nl + 16*sub)
#pragma unroll
      for (int sub = 0; sub < 2; ++sub) {
        const int kcol = k0 + sub * 16 + nl;
#pragma unroll
        for (int r = 0; r < 8; ++r) {
          const int qrow = q0 + r + 8 * half;
          if (kcol > qrow) S[sub][r] = -3.0e38f;
        }
      }
#pragma unroll
      for (int r = 0; r < 8; ++r) {
        float mx = fmaxf(S[0][r], S[1][r]);
#pragma unroll
        for (int off = 8; off >= 1; off >>= 1)
          mx = fmaxf(mx, __shfl_xor(mx, off, 32));   // reduce within 16-lane half
        const float m_new = fmaxf(m_run[r], mx);
        const float alpha = __expf(m_run[r] - m_new);
        const float p0 = __expf(S[0][r] - m_new);
        const float p1 = __expf(S[1][r] - m_new);
        S[0][r] = p0;
        S[1][r] = p1;
        float ps = p0 + p1;
#pragma unroll
        for (int off = 8; off >= 1; off >>= 1)
          ps += __shfl_xor(ps, off, 32);
        l_run[r] = l_run[r] * alpha + ps;
        m_run[r] = m_new;
#pragma unroll
        for (int t = 0; t < 4; ++t) O[t][r] *= alpha;
      }

      // C-layout P -> row-major bf16 in per-wave LDS (same-wave DS ops are
      // in-order), then reload as A-fragment for the PV matmul.
      __bf16* pb = Pbuf[wave];
#pragma unroll
      for (int sub = 0; sub < 2; ++sub)
#pragma unroll
        for (int r = 0; r < 8; ++r)
          pb[(r + 8 * half) * 32 + sub * 16 + nl] = (__bf16)S[sub][r];

      v16bf pa;
      {
        const __bf16* p = pb + nl * 32 + 16 * half;
        pa = cat8(*(const v8bf*)p, *(const v8bf*)(p + 8));
      }

      // O += P V : 4 N-tiles over HEAD, K = 32 keys.
#pragma unroll
      for (int t = 0; t < 4; ++t) {
        const __bf16* vp = &Vt[(16 * t + nl) * 32 + 16 * half];
        v16bf bf = *(const v16bf*)vp;
        O[t] = __builtin_amdgcn_wmma_f32_16x16x32_bf16(false, pa, false, bf, (short)0, O[t], false, false);
      }
    }
  }

  // Normalize and write fp32 output (B_, T_, H_).
#pragma unroll
  for (int r = 0; r < 8; ++r) {
    const float inv = 1.0f / l_run[r];
    const int row = base + q0 + r + 8 * half;
#pragma unroll
    for (int t = 0; t < 4; ++t)
      out[row * H_ + 16 * t + nl] = O[t][r] * inv;
  }
}

// ---------------------------------------------------------------------------
extern "C" void kernel_launch(void* const* d_in, const int* in_sizes, int n_in,
                              void* d_out, int out_size, void* d_ws, size_t ws_size,
                              hipStream_t stream) {
  const float* x    = (const float*)d_in[0];
  const float* cosb = (const float*)d_in[1];
  const float* sinb = (const float*)d_in[2];
  // d_in[3] = tril : causal mask handled analytically, unused.
  const float* Wq   = (const float*)d_in[4];
  const float* Wk   = (const float*)d_in[5];
  const float* Wv   = (const float*)d_in[6];
  float* out = (float*)d_out;

  // Workspace layout (bf16): Wt_q/k/v (64x1024 each), then Q/K/V (16384x64 each).
  __bf16* Wtq = (__bf16*)d_ws;
  __bf16* Wtk = Wtq + H_ * C_;
  __bf16* Wtv = Wtk + H_ * C_;
  __bf16* Qb  = Wtv + H_ * C_;
  __bf16* Kb  = Qb + (size_t)B_ * T_ * H_;
  __bf16* Vb  = Kb + (size_t)B_ * T_ * H_;

  transpose_w_kernel<<<(C_ * H_) / 256, 256, 0, stream>>>(Wq, Wtq);
  transpose_w_kernel<<<(C_ * H_) / 256, 256, 0, stream>>>(Wk, Wtk);
  transpose_w_kernel<<<(C_ * H_) / 256, 256, 0, stream>>>(Wv, Wtv);

  qkv_rope_kernel<<<(B_ * T_) / 128, 256, 0, stream>>>(x, cosb, sinb, Wtq, Wtk, Wtv,
                                                       Qb, Kb, Vb);

  attn_kernel<<<dim3(T_ / 64, B_), 128, 0, stream>>>(Qb, Kb, Vb, out);
}
